// CupyNet_29257317221021
// MI455X (gfx1250) — matrix operand order repaired and checked by reading, not compiled, not run on hardware
//
#include <hip/hip_runtime.h>

// ---------------------------------------------------------------------------
// Spiking CNN (IF neurons, T=8) for MI455X / gfx1250, wave32 + WMMA f16.
// conv2 (implicit GEMM, 59 GF) and fc1 (26 GF) run on v_wmma_f32_16x16x32_f16
// with fp32 accumulate. Spikes are exactly {0,1} -> lossless in fp16.
// Everything fits in the 192MB L2, so we optimize for WMMA issue rate, not HBM.
// Workspace layout (region reuse), peak ~180.2 MB:
//   [0          ..  51,380,224) m1   : pooled spikes  [TB,14,14,128] f16 (NHWC)
//                                      reused later by zfc [TB,6272] f16
//   [51,380,224 .. 154,140,672) z2   : conv2+bn2 out  [TB,14,14,128] f32 (NHWC)
//                                      reused later by H [TB,2048] f32 + s3 f16
//   [154,140,672.. 154,435,584) w2t  : conv2 weights  [9][128o][128c] f16
//   [154,435,584.. 180,125,696) f1h  : fc1 weights    [2048][6272] f16
// ---------------------------------------------------------------------------

typedef __attribute__((ext_vector_type(16))) _Float16 v16h;
typedef __attribute__((ext_vector_type(8)))  _Float16 v8h;
typedef __attribute__((ext_vector_type(8)))  float    v8f;

union F16x16 { v16h v; v8h h[2]; };

__device__ __forceinline__ v8f wmma16x16x32(v16h a, v16h b, v8f c) {
    // 8 args: (neg_a, A, neg_b, B, c_mod, C, reuse_a, reuse_b)
    return __builtin_amdgcn_wmma_f32_16x16x32_f16(false, a, false, b,
                                                  (short)0, c, false, false);
}

#define V_TH 1.0f
#define EPS  1e-5f

// ---------------------------------------------------------------------------
// K1: conv1(1->128, 3x3 SAME, 28x28) + bn1 + IF(T=8, constant drive) + 2x2 pool
//     -> m1[t*128+b][h2][w2][c] f16 in {0,1}.  One thread per (b,c,h2,w2).
// ---------------------------------------------------------------------------
__global__ __launch_bounds__(256) void
k_conv1_if_pool(const float* __restrict__ x, const float* __restrict__ w1,
                const float* __restrict__ g, const float* __restrict__ be,
                const float* __restrict__ mu, const float* __restrict__ va,
                _Float16* __restrict__ m1)
{
    int tid = blockIdx.x * blockDim.x + threadIdx.x;   // ((b*14+h2)*14+w2)*128+c
    if (tid >= 128 * 14 * 14 * 128) return;
    int c   = tid & 127;
    int r   = tid >> 7;
    int w2  = r % 14;
    int h2  = (r / 14) % 14;
    int b   = r / 196;

    float inv  = g[c] * rsqrtf(va[c] + EPS);
    float beta = be[c] - mu[c] * inv;

    float wk[9];
#pragma unroll
    for (int i = 0; i < 9; ++i) wk[i] = w1[c * 9 + i];

    float y[4];
#pragma unroll
    for (int d = 0; d < 4; ++d) {
        int h = 2 * h2 + (d >> 1), w = 2 * w2 + (d & 1);
        float acc = 0.f;
#pragma unroll
        for (int kh = 0; kh < 3; ++kh)
#pragma unroll
            for (int kw = 0; kw < 3; ++kw) {
                int hh = h + kh - 1, ww = w + kw - 1;
                if (hh >= 0 && hh < 28 && ww >= 0 && ww < 28)
                    acc += x[(b * 28 + hh) * 28 + ww] * wk[kh * 3 + kw];
            }
        y[d] = acc * inv + beta;
    }

    float v4[4] = {0.f, 0.f, 0.f, 0.f};
#pragma unroll
    for (int t = 0; t < 8; ++t) {
        float smax = 0.f;
#pragma unroll
        for (int d = 0; d < 4; ++d) {
            float hc = v4[d] + y[d];
            float s  = (hc >= V_TH) ? 1.f : 0.f;
            v4[d]    = hc * (1.f - s);           // hard reset
            smax     = fmaxf(smax, s);
        }
        m1[((((size_t)(t * 128 + b)) * 14 + h2) * 14 + w2) * 128 + c] = (_Float16)smax;
    }
}

// ---------------------------------------------------------------------------
// K2: conv2_w [o][c][3][3] f32 -> w2t [tap][o][c] f16 (B column-major per tap)
// ---------------------------------------------------------------------------
__global__ __launch_bounds__(256) void
k_w2_transform(const float* __restrict__ w2, _Float16* __restrict__ w2t)
{
    int tid = blockIdx.x * blockDim.x + threadIdx.x;
    if (tid >= 128 * 128 * 9) return;
    int tap = tid % 9;
    int c   = (tid / 9) & 127;
    int o   = tid / (9 * 128);
    w2t[((size_t)tap * 128 + o) * 128 + c] = (_Float16)w2[(size_t)(o * 128 + c) * 9 + tap];
}

// K3: fp32 -> fp16 copy (fc1 weights; already n-major over k)
__global__ __launch_bounds__(256) void
k_cvt_half(const float* __restrict__ src, _Float16* __restrict__ dst, int n)
{
    int i = blockIdx.x * blockDim.x + threadIdx.x;
    if (i < n) dst[i] = (_Float16)src[i];
}

// ---------------------------------------------------------------------------
// K4: conv2 as implicit GEMM on WMMA.  M = pixels (16/tile, 13 tiles/image),
// N = 64 out-ch per wave, K = 9 taps x 128 in-ch (36 k-steps of 32).
// A-frag: lane row m=l&15 -> pixel; 16 contiguous K at lh*16 (matches B).
// B-frag: lane col n=l&15; 16 contiguous K at lh*16 (w2t is [tap][o][c]).
// bn2 fused on accumulators.  Wave count: 1024 * 13 * 2 = 26624.
// ---------------------------------------------------------------------------
__global__ __launch_bounds__(128) void
k_conv2_wmma(const _Float16* __restrict__ m1, const _Float16* __restrict__ w2t,
             const float* __restrict__ g, const float* __restrict__ be,
             const float* __restrict__ mu, const float* __restrict__ va,
             float* __restrict__ z2)
{
    int wv   = (blockIdx.x * blockDim.x + threadIdx.x) >> 5;
    int lane = threadIdx.x & 31;
    if (wv >= 1024 * 13 * 2) return;
    int ntile = wv & 1;              // 2 tiles of 64 out-channels
    int ptile = (wv >> 1) % 13;      // 13 tiles of 16 pixels (196 -> pad 208)
    int tb    = (wv >> 1) / 13;      // fused (t,b) image index
    int nbase = ntile * 64;
    int lm = lane & 15, lh = lane >> 4;

    int p  = ptile * 16 + lm;        // this lane's A row = pixel
    int ph = p / 14, pw = p % 14;
    bool pvalid = (p < 196);

    const _Float16* blane = w2t + ((size_t)(nbase + lm)) * 128 + lh * 16;

    v8f acc[4] = {};
    for (int tap = 0; tap < 9; ++tap) {
        int kh = tap / 3, kw = tap % 3;
        int hh = ph + kh - 1, ww = pw + kw - 1;
        bool tv = pvalid && hh >= 0 && hh < 14 && ww >= 0 && ww < 14;
        const _Float16* abase = m1 + (((size_t)tb * 14 + hh) * 14 + ww) * 128;
        const _Float16* bbase = blane + (size_t)tap * 128 * 128;
#pragma unroll
        for (int ks = 0; ks < 4; ++ks) {          // 4 x K=32 over 128 channels
            int c0 = ks * 32 + lh * 16;
            F16x16 a;
            if (tv) {
                a.h[0] = *(const v8h*)(abase + c0);
                a.h[1] = *(const v8h*)(abase + c0 + 8);
            } else {
                a.h[0] = (v8h){};
                a.h[1] = (v8h){};
            }
            F16x16 b0, b1, b2, b3;
            b0.v = *(const v16h*)(bbase + ks * 32);
            b1.v = *(const v16h*)(bbase + 16 * 128 + ks * 32);
            b2.v = *(const v16h*)(bbase + 32 * 128 + ks * 32);
            b3.v = *(const v16h*)(bbase + 48 * 128 + ks * 32);
            acc[0] = wmma16x16x32(a.v, b0.v, acc[0]);
            acc[1] = wmma16x16x32(a.v, b1.v, acc[1]);
            acc[2] = wmma16x16x32(a.v, b2.v, acc[2]);
            acc[3] = wmma16x16x32(a.v, b3.v, acc[3]);
        }
    }

    // C layout: lane col n = lm, rows m = j + 8*lh.  Fuse bn2, store NHWC f32.
#pragma unroll
    for (int s = 0; s < 4; ++s) {
        int n = nbase + s * 16 + lm;
        float inv  = g[n] * rsqrtf(va[n] + EPS);
        float beta = be[n] - mu[n] * inv;
#pragma unroll
        for (int j = 0; j < 8; ++j) {
            int pp = ptile * 16 + j + 8 * lh;
            if (pp < 196) {
                z2[(((size_t)tb * 14 + pp / 14) * 14 + pp % 14) * 128 + n] =
                    acc[s][j] * inv + beta;
            }
        }
    }
}

// ---------------------------------------------------------------------------
// K5: IF(T=8, sequential state) on z2 + 2x2 pool + NCHW flatten -> zfc f16
//     zfc[(t*128+b)][c*49 + h2*7 + w2]  (matches fc1_w's i ordering)
// ---------------------------------------------------------------------------
__global__ __launch_bounds__(256) void
k_if2_pool(const float* __restrict__ z2, _Float16* __restrict__ zfc)
{
    int tid = blockIdx.x * blockDim.x + threadIdx.x;   // ((b*7+h2)*7+w2)*128+c
    if (tid >= 128 * 7 * 7 * 128) return;
    int c  = tid & 127;
    int r  = tid >> 7;
    int w2 = r % 7;
    int h2 = (r / 7) % 7;
    int b  = r / 49;

    float vs[4] = {0.f, 0.f, 0.f, 0.f};
#pragma unroll
    for (int t = 0; t < 8; ++t) {
        float smax = 0.f;
#pragma unroll
        for (int d = 0; d < 4; ++d) {
            int hh = 2 * h2 + (d >> 1), ww = 2 * w2 + (d & 1);
            float zz = z2[(((size_t)(t * 128 + b) * 14 + hh) * 14 + ww) * 128 + c];
            float hc = vs[d] + zz;
            float s  = (hc >= V_TH) ? 1.f : 0.f;
            vs[d]    = hc * (1.f - s);
            smax     = fmaxf(smax, s);
        }
        zfc[(size_t)(t * 128 + b) * 6272 + c * 49 + h2 * 7 + w2] = (_Float16)smax;
    }
}

// ---------------------------------------------------------------------------
// K6: fc1 GEMM on WMMA.  H[M=1024][N=2048] = zfc[M][K=6272] x f1h[N][K]^T.
// Wave tile 16x64, 196 k-steps of 32.  64 M-tiles x 32 N-tiles = 2048 waves.
// ---------------------------------------------------------------------------
__global__ __launch_bounds__(128) void
k_fc1_wmma(const _Float16* __restrict__ A, const _Float16* __restrict__ Bw,
           float* __restrict__ H)
{
    const int K = 6272;
    int wv   = (blockIdx.x * blockDim.x + threadIdx.x) >> 5;
    int lane = threadIdx.x & 31;
    if (wv >= 64 * 32) return;
    int nt = wv & 31;          // N tile of 64
    int mt = wv >> 5;          // M tile of 16
    int lm = lane & 15, lh = lane >> 4;

    const _Float16* ap = A + (size_t)(mt * 16 + lm) * K + lh * 16;
    const _Float16* bp = Bw + (size_t)(nt * 64 + lm) * K + lh * 16;

    v8f acc[4] = {};
    for (int ks = 0; ks < 196; ++ks) {
        int k0 = ks * 32;
        F16x16 a, b0, b1, b2, b3;
        a.h[0] = *(const v8h*)(ap + k0);
        a.h[1] = *(const v8h*)(ap + k0 + 8);
        b0.v = *(const v16h*)(bp + k0);
        b1.v = *(const v16h*)(bp + (size_t)16 * K + k0);
        b2.v = *(const v16h*)(bp + (size_t)32 * K + k0);
        b3.v = *(const v16h*)(bp + (size_t)48 * K + k0);
        __builtin_prefetch(bp + k0 + 256, 0, 0);   // global_prefetch_b8
        acc[0] = wmma16x16x32(a.v, b0.v, acc[0]);
        acc[1] = wmma16x16x32(a.v, b1.v, acc[1]);
        acc[2] = wmma16x16x32(a.v, b2.v, acc[2]);
        acc[3] = wmma16x16x32(a.v, b3.v, acc[3]);
    }

#pragma unroll
    for (int s = 0; s < 4; ++s) {
        int n = nt * 64 + s * 16 + lm;
#pragma unroll
        for (int j = 0; j < 8; ++j) {
            int m = mt * 16 + j + 8 * lh;
            H[(size_t)m * 2048 + n] = acc[s][j];
        }
    }
}

// K7: IF over T on fc1 pre-activations -> s3 f16 in {0,1}
__global__ __launch_bounds__(256) void
k_if3(const float* __restrict__ H, _Float16* __restrict__ s3)
{
    int tid = blockIdx.x * blockDim.x + threadIdx.x;   // b*2048 + i
    if (tid >= 128 * 2048) return;
    float vv = 0.f;
#pragma unroll
    for (int t = 0; t < 8; ++t) {
        size_t idx = (size_t)t * 128 * 2048 + tid;
        float hc = vv + H[idx];
        float s  = (hc >= V_TH) ? 1.f : 0.f;
        vv = hc * (1.f - s);
        s3[idx] = (_Float16)s;
    }
}

// K8: fc2 (2048 -> 10) + IF + mean over T.  Tiny (42 MF): scalar dot per (b,j).
__global__ __launch_bounds__(256) void
k_fc2_if_mean(const _Float16* __restrict__ s3, const float* __restrict__ w,
              float* __restrict__ out)
{
    int tid = blockIdx.x * blockDim.x + threadIdx.x;
    if (tid >= 128 * 10) return;
    int j = tid % 10, b = tid / 10;
    float ot[8];
#pragma unroll
    for (int t = 0; t < 8; ++t) {
        const _Float16* sp = s3 + (size_t)(t * 128 + b) * 2048;
        const float*    wp = w + (size_t)j * 2048;
        float acc = 0.f;
        for (int i = 0; i < 2048; ++i) acc += (float)sp[i] * wp[i];
        ot[t] = acc;
    }
    float vv = 0.f, sum = 0.f;
#pragma unroll
    for (int t = 0; t < 8; ++t) {
        float hc = vv + ot[t];
        float s  = (hc >= V_TH) ? 1.f : 0.f;
        vv  = hc * (1.f - s);
        sum += s;
    }
    out[b * 10 + j] = sum * 0.125f;
}

// ---------------------------------------------------------------------------
extern "C" void kernel_launch(void* const* d_in, const int* in_sizes, int n_in,
                              void* d_out, int out_size, void* d_ws, size_t ws_size,
                              hipStream_t stream)
{
    (void)in_sizes; (void)n_in; (void)out_size; (void)ws_size;
    const float* x   = (const float*)d_in[0];
    const float* c1w = (const float*)d_in[1];
    const float* g1  = (const float*)d_in[2];
    const float* be1 = (const float*)d_in[3];
    const float* mu1 = (const float*)d_in[4];
    const float* va1 = (const float*)d_in[5];
    const float* c2w = (const float*)d_in[6];
    const float* g2  = (const float*)d_in[7];
    const float* be2 = (const float*)d_in[8];
    const float* mu2 = (const float*)d_in[9];
    const float* va2 = (const float*)d_in[10];
    const float* f1w = (const float*)d_in[11];
    const float* f2w = (const float*)d_in[12];
    // d_in[13] = T, fixed at 8.

    char* ws = (char*)d_ws;                       // needs >= ~180.2 MB
    const size_t OFF_M1  = 0;
    const size_t OFF_Z2  = 51380224;
    const size_t OFF_W2T = OFF_Z2 + 102760448;    // 154,140,672
    const size_t OFF_F1W = OFF_W2T + 294912;      // 154,435,584
    const size_t OFF_ZFC = 0;                     // reuses m1 region
    const size_t OFF_H   = OFF_Z2;                // reuses z2 region
    const size_t OFF_S3  = OFF_Z2 + 8388608;      // after H in z2 region

    _Float16* m1p = (_Float16*)(ws + OFF_M1);
    float*    z2p = (float*)(ws + OFF_Z2);
    _Float16* w2t = (_Float16*)(ws + OFF_W2T);
    _Float16* f1h = (_Float16*)(ws + OFF_F1W);
    _Float16* zfc = (_Float16*)(ws + OFF_ZFC);
    float*    Hp  = (float*)(ws + OFF_H);
    _Float16* s3p = (_Float16*)(ws + OFF_S3);
    float*    out = (float*)d_out;

    k_conv1_if_pool<<<12544, 256, 0, stream>>>(x, c1w, g1, be1, mu1, va1, m1p);
    k_w2_transform <<<576,   256, 0, stream>>>(c2w, w2t);
    k_cvt_half     <<<50176, 256, 0, stream>>>(f1w, f1h, 2048 * 6272);
    k_conv2_wmma   <<<6656,  128, 0, stream>>>(m1p, w2t, g2, be2, mu2, va2, z2p);
    k_if2_pool     <<<3136,  256, 0, stream>>>(z2p, zfc);
    k_fc1_wmma     <<<512,   128, 0, stream>>>(zfc, f1h, Hp);
    k_if3          <<<1024,  256, 0, stream>>>(Hp, s3p);
    k_fc2_if_mean  <<<5,     256, 0, stream>>>(s3p, f2w, out);
}
